// DenseSIFTDescriptor_1726576854302
// MI455X (gfx1250) — matrix operand order repaired and checked by reading, not compiled, not run on hardware
//
#include <hip/hip_runtime.h>

typedef __attribute__((ext_vector_type(16))) _Float16 v16h;
typedef __attribute__((ext_vector_type(8)))  _Float16 v8h;
typedef __attribute__((ext_vector_type(8)))  float    v8f;

#define HW   384
#define TILE 16
#define TWO_PI 6.28318530717958647692f

// Fused DenseSIFT: gradients -> 8-bin soft histogram -> separable 4-tap
// triangular pooling as banded-Toeplitz WMMA matmuls (operands pre-swizzled
// in LDS for contiguous b128 loads) -> descriptor gather + L2/clip/L2/
// RootSIFT normalization -> coalesced non-temporal stores.
__global__ __launch_bounds__(256) void dsift_kernel(const float* __restrict__ x,
                                                    const float* __restrict__ pool_w,
                                                    float* __restrict__ out)
{
    // LDS working set (~50 KB)
    __shared__ __align__(16) float    xs[24][25];        // x tile + halo (replicate-clamped)
    __shared__ __align__(32) _Float16 histW[8][32][32];  // histogram, WMMA-A K-permuted, zero-padded
    __shared__ __align__(32) _Float16 tmpT[8][32][32];   // horiz-pooled, transposed [col][row]
    __shared__ __align__(16) float    angb[8][19][20];   // fully pooled ang tile (f32)
    __shared__ __align__(32) _Float16 tblBH[2][32][16];  // pass-H B operand (Toeplitz), per-lane
    __shared__ __align__(32) _Float16 tblAV[2][32][16];  // pass-V A operand (Toeplitz), per-lane

    const int tid = threadIdx.x;
    const int gx0 = blockIdx.x * TILE;
    const int gy0 = blockIdx.y * TILE;
    const int b   = blockIdx.z;

    // Separable 1-D pooling weights recovered from pool_w = outer(w1, w1)
    const float w1f = sqrtf(pool_w[5]);            // w1[1]
    const float w0f = pool_w[4] / w1f;             // w1[0]
    const float w2f = pool_w[6] / w1f;             // w1[2]
    const float w3f = pool_w[7] / w1f;             // w1[3]

    // ---- Phase 1a: zero histW (K/M tail must be exact 0 for unguarded WMMA A loads)
    for (int idx = tid; idx < 2048; idx += 256)
        ((unsigned long long*)histW)[idx] = 0ull;

    // ---- Phase 1b: load x tile (rows/cols gy0-4 .. gy0+19, replicate pad) ----
    for (int idx = tid; idx < 24 * 24; idx += 256) {
        int r = idx / 24, c = idx % 24;
        int py = gy0 + r - 4; py = py < 0 ? 0 : (py > HW - 1 ? HW - 1 : py);
        int px = gx0 + c - 4; px = px < 0 ? 0 : (px > HW - 1 ? HW - 1 : px);
        xs[r][c] = x[((size_t)b * HW + py) * HW + px];
    }

    // ---- Phase 1c: build Toeplitz operand tables (per-lane contiguous) ----
    // tblBH: pass-H B, B[k,t] = w1[k-t] (t = output col, k = input col), 16-bit B K-map
    for (int idx = tid; idx < 1024; idx += 256) {
        int nb = idx >> 9, lane = (idx >> 4) & 31, e = idx & 15;
        int half = lane >> 4, lm = lane & 15;
        int t = nb * 16 + lm;
        int k = (e & 7) + (half << 3) + ((e >> 3) << 4);
        int dlt = k - t;
        float w = (dlt == 0) ? w0f : (dlt == 1) ? w1f : (dlt == 2) ? w2f : (dlt == 3) ? w3f : 0.0f;
        if (t >= 19) w = 0.0f;
        tblBH[nb][lane][e] = (_Float16)w;
    }
    // tblAV: pass-V A, A[a,k] = w1[k-a] (a = ang row, k = tmp row), contiguous K-map (k = 16*half + e)
    for (int idx = tid; idx < 1024; idx += 256) {
        int mb = idx >> 9, lane = (idx >> 4) & 31, e = idx & 15;
        int half = lane >> 4, lm = lane & 15;
        int a = mb * 16 + lm;
        int k = (half << 4) + e;
        int dlt = k - a;
        float w = (dlt == 0) ? w0f : (dlt == 1) ? w1f : (dlt == 2) ? w2f : (dlt == 3) ? w3f : 0.0f;
        if (a >= 19) w = 0.0f;
        tblAV[mb][lane][e] = (_Float16)w;
    }
    __syncthreads();

    // ---- Phase 2: soft orientation histogram, 22x22 region (global -3..18) ----
    // Stored with K-permuted column index so WMMA-A operands are contiguous.
    for (int idx = tid; idx < 22 * 22; idx += 256) {
        int r = idx / 22, c = idx % 22;
        int py = gy0 + r - 3, px = gx0 + c - 3;
        float w0 = 0.0f, w1 = 0.0f; int bo0 = 0, bo1 = 1;
        if (py >= 0 && py < HW && px >= 0 && px < HW) {   // hist is zero off-image
            float gxv = (xs[r + 1][c + 2] - xs[r + 1][c]) * 0.5f;
            float gyv = (xs[r + 2][c + 1] - xs[r][c + 1]) * 0.5f;
            float mag = sqrtf(gxv * gxv + gyv * gyv + 1e-10f);
            float ori = atan2f(gyv, gxv + 1e-10f) + TWO_PI;
            float obig = ori * (8.0f / TWO_PI);           // in (4, 12]
            float b0f  = floorf(obig);
            float wo1  = obig - b0f;
            bo0 = ((int)b0f) & 7;
            bo1 = (bo0 + 1) & 7;
            w0 = (1.0f - wo1) * mag;
            w1 = wo1 * mag;
        }
        int cp = (((c >> 3) & 1) << 4) + (c & 7) + ((c >> 4) << 3);  // A K-permutation
        #pragma unroll
        for (int d = 0; d < 8; ++d) {
            float v = (d == bo0) ? w0 : ((d == bo1) ? w1 : 0.0f);
            histW[d][r][cp] = (_Float16)v;
        }
    }
    __syncthreads();

    const int wave = tid >> 5;      // one channel per wave (8 waves, 8 channels)
    const int lane = tid & 31;
    const int half = lane >> 4;
    const int lm   = lane & 15;

    // ---- Phase 3: horizontal pooling via WMMA: tmp[row][t] = sum_v w1[v]*hist[row][t+v]
    #pragma unroll
    for (int mb = 0; mb < 2; ++mb) {
        const int row = mb * 16 + lm;                 // rows 22..31 are zero-filled
        const v16h A = *(const v16h*)&histW[wave][row][half << 4];
        #pragma unroll
        for (int nb = 0; nb < 2; ++nb) {
            const v16h Bm = *(const v16h*)&tblBH[nb][lane][0];
            v8f D = {0.f, 0.f, 0.f, 0.f, 0.f, 0.f, 0.f, 0.f};
            D = __builtin_amdgcn_wmma_f32_16x16x32_f16(false, A, false, Bm,
                                                       (short)0, D, false, false);
            // D rows (mb*16 + half*8 + r8) are 8 consecutive -> one packed 16B store,
            // transposed so pass-V B operands are contiguous. All 32x32 written; the
            // out-of-band entries are finite and get multiplied by zero weights.
            v8h dh;
            #pragma unroll
            for (int r8 = 0; r8 < 8; ++r8) dh[r8] = (_Float16)D[r8];
            const int ocol = nb * 16 + lm;
            *(v8h*)&tmpT[wave][ocol][mb * 16 + (half << 3)] = dh;
        }
    }
    __syncthreads();

    // ---- Phase 4: vertical pooling via WMMA: ang[a][t] = sum_u w1[u]*tmp[a+u][t]
    #pragma unroll
    for (int mb = 0; mb < 2; ++mb) {
        const v16h A = *(const v16h*)&tblAV[mb][lane][0];
        #pragma unroll
        for (int nb = 0; nb < 2; ++nb) {
            const int col = nb * 16 + lm;
            const v16h Bm = *(const v16h*)&tmpT[wave][col][half << 4];
            v8f D = {0.f, 0.f, 0.f, 0.f, 0.f, 0.f, 0.f, 0.f};
            D = __builtin_amdgcn_wmma_f32_16x16x32_f16(false, A, false, Bm,
                                                       (short)0, D, false, false);
            #pragma unroll
            for (int r8 = 0; r8 < 8; ++r8) {
                int orow = mb * 16 + (half << 3) + r8;
                int ocol = col;
                if (orow < 19 && ocol < 19) {
                    int ay = gy0 + orow - 1;            // ang global index, valid [0,384]
                    int ax = gx0 + ocol - 1;
                    float v = D[r8];
                    if (ay < 0 || ay > HW || ax < 0 || ax > HW) v = 0.0f; // zero-pad of 2nd conv
                    angb[wave][orow][ocol] = v;
                }
            }
        }
    }
    __syncthreads();

    // ---- Phase 5: descriptor gather + L2 / clip(0.2) / L2 / RootSIFT, NT stores ----
    const int i = tid >> 4, j = tid & 15;      // output pixel within tile

    float s2 = 0.0f;
    #pragma unroll
    for (int d = 0; d < 8; ++d)
        #pragma unroll
        for (int ky = 0; ky < 4; ++ky)
            #pragma unroll
            for (int kx = 0; kx < 4; ++kx) {
                float v = angb[d][i + ky][j + kx];
                s2 += v * v;
            }
    const float inva = 1.0f / fmaxf(sqrtf(s2), 1e-12f);

    float s2b = 0.0f, s1b = 0.0f;
    #pragma unroll
    for (int d = 0; d < 8; ++d)
        #pragma unroll
        for (int ky = 0; ky < 4; ++ky)
            #pragma unroll
            for (int kx = 0; kx < 4; ++kx) {
                float v = fminf(fmaxf(angb[d][i + ky][j + kx], 0.0f) * inva, 0.2f);
                s2b += v * v;
                s1b += v;
            }
    const float bn    = fmaxf(sqrtf(s2b), 1e-12f);
    const float n1    = s1b / bn;
    const float scale = 1.0f / (bn * fmaxf(n1, 1e-12f));

    const size_t plane = (size_t)HW * HW;
    const size_t base  = (size_t)b * 128 * plane + (size_t)(gy0 + i) * HW + (size_t)(gx0 + j);
    #pragma unroll
    for (int d = 0; d < 8; ++d)
        #pragma unroll
        for (int ky = 0; ky < 4; ++ky)
            #pragma unroll
            for (int kx = 0; kx < 4; ++kx) {
                int cch = d * 16 + ky * 4 + kx;
                float v = fminf(fmaxf(angb[d][i + ky][j + kx], 0.0f) * inva, 0.2f);
                float o = sqrtf(v * scale + 1e-10f);
                __builtin_nontemporal_store(o, out + base + (size_t)cch * plane);
            }
}

extern "C" void kernel_launch(void* const* d_in, const int* in_sizes, int n_in,
                              void* d_out, int out_size, void* d_ws, size_t ws_size,
                              hipStream_t stream) {
    (void)in_sizes; (void)n_in; (void)out_size; (void)d_ws; (void)ws_size;
    const float* x      = (const float*)d_in[0];   // (4,1,384,384) f32
    const float* pool_w = (const float*)d_in[1];   // (4,4) f32
    // d_in[2] (reshape_w) is the identity-reshape kernel; folded analytically.
    float* out = (float*)d_out;                    // (4,128,384,384) f32

    dim3 grid(HW / TILE, HW / TILE, 4);            // 24 x 24 x 4 tiles
    dim3 block(256);                               // 8 wave32s; one channel per wave
    dsift_kernel<<<grid, block, 0, stream>>>(x, pool_w, out);
}
